// BSFSNet_71451075937139
// MI455X (gfx1250) — matrix operand order repaired and checked by reading, not compiled, not run on hardware
//
#include <hip/hip_runtime.h>
#include <hip/hip_bf16.h>
#include <stdint.h>

// ---- constants from the reference -----------------------------------------
#define B_SZ   1024
#define F_SZ   1024
#define H_SZ   128
#define C_SZ   100
#define KFC    8
#define KSB    4
#define NSEL   (KFC * F_SZ)          // 8192
#define NROWS  (B_SZ * KFC * KSB)    // 32768

typedef __attribute__((ext_vector_type(16))) __bf16 bf16x16;
typedef __attribute__((ext_vector_type(8)))  float  f32x8;

union AFrag { bf16x16 v; unsigned u[8]; __bf16 h[16]; };
union Pack4 { __bf16 h[4]; uint2 u2; };

__device__ __forceinline__ __bf16 to_bf16(float f) { return (__bf16)f; }

// ---- CDNA5 async global->LDS path (guarded; falls back to sync staging) ----
#if defined(__has_builtin)
#  if __has_builtin(__builtin_amdgcn_global_load_async_to_lds_b128) && \
      __has_builtin(__builtin_amdgcn_s_wait_asynccnt)
#    define HAS_ASYNC 1
#  endif
#endif
#ifndef HAS_ASYNC
#  define HAS_ASYNC 0
#endif

#if HAS_ASYNC
// builtin signature (from probe diagnostic): param0 is int4 (vector_size 16) in AS1
typedef int v4i_async __attribute__((vector_size(16)));
typedef __attribute__((address_space(1))) v4i_async glb_v4i_t;
typedef __attribute__((address_space(3))) v4i_async lds_v4i_t;
// generic->AS cast via integer: flat LDS addr low 32 bits == LDS offset (ISA 10.2)
__device__ __forceinline__ glb_v4i_t* as_g128(const void* p) {
    return (glb_v4i_t*)(uintptr_t)p;
}
__device__ __forceinline__ lds_v4i_t* as_l128(void* p) {
    return (lds_v4i_t*)(uintptr_t)p;
}
#endif

// ---------------------------------------------------------------------------
// Kernel 1: S = x @ W_s + b_s   (1024x1024)*(1024x8192), bf16 WMMA, f32 acc
// block = 256 threads (8 waves). Tile: 128 rows x 64 cols. K-step = 32.
// A tile staged fp32 (async DMA when available, double-buffered), converted
// to bf16 at fragment build; B tile staged bf16 (each element read by 8 waves).
// ---------------------------------------------------------------------------
__device__ __forceinline__ void sel_stage_A(const float* __restrict__ x, int m0,
                                            int k0, float* buf, int tid)
{
#if HAS_ASYNC
    #pragma unroll
    for (int i = 0; i < 4; ++i) {
        int f4 = tid + i * 256;                 // 0..1023 float4s of 128x32 tile
        int r = f4 >> 3, kq = f4 & 7;
        const float* g = &x[(m0 + r) * F_SZ + k0 + kq * 4];
        float* l = &buf[r * 32 + kq * 4];
        __builtin_amdgcn_global_load_async_to_lds_b128(as_g128(g), as_l128(l), 0, 0);
    }
#else
    #pragma unroll
    for (int i = 0; i < 4; ++i) {
        int f4 = tid + i * 256;
        int r = f4 >> 3, kq = f4 & 7;
        float4 v = *(const float4*)&x[(m0 + r) * F_SZ + k0 + kq * 4];
        *(float4*)&buf[r * 32 + kq * 4] = v;
    }
#endif
}

__device__ __forceinline__ void sel_load_B(const float* __restrict__ Ws, int n0,
                                           int k0, float4* br, int tid)
{
    #pragma unroll
    for (int i = 0; i < 2; ++i) {
        int f4 = tid + i * 256;                 // 0..511 float4s of 32x64 tile
        int k = f4 >> 4, nq = f4 & 15;
        br[i] = *(const float4*)&Ws[(k0 + k) * NSEL + n0 + nq * 4];
    }
}

__device__ __forceinline__ void sel_store_B(__bf16* Bsb, const float4* br, int tid)
{
    #pragma unroll
    for (int i = 0; i < 2; ++i) {
        int f4 = tid + i * 256;
        int k = f4 >> 4, nq = f4 & 15;
        Bsb[(nq * 4 + 0) * 32 + k] = to_bf16(br[i].x);
        Bsb[(nq * 4 + 1) * 32 + k] = to_bf16(br[i].y);
        Bsb[(nq * 4 + 2) * 32 + k] = to_bf16(br[i].z);
        Bsb[(nq * 4 + 3) * 32 + k] = to_bf16(br[i].w);
    }
}

__global__ __launch_bounds__(256)
void selector_gemm(const float* __restrict__ x, const float* __restrict__ Ws,
                   const float* __restrict__ bs, float* __restrict__ S)
{
    __shared__ alignas(16) float  Asf[2][128][32];   // fp32 (row, k), double-buffered
    __shared__ alignas(16) __bf16 Bs[2][64][32];     // bf16 (col, k), double-buffered

    const int tid  = threadIdx.x;
    const int lane = tid & 31;
    const int wv   = tid >> 5;
    const int n0   = blockIdx.x * 64;
    const int m0   = blockIdx.y * 128;

    // prologue: stage buffer 0
    sel_stage_A(x, m0, 0, &Asf[0][0][0], tid);
    {
        float4 br[2];
        sel_load_B(Ws, n0, 0, br, tid);
        sel_store_B(&Bs[0][0][0], br, tid);
    }
#if HAS_ASYNC
    __builtin_amdgcn_s_wait_asynccnt(0);
#endif
    __syncthreads();

    f32x8 acc[4] = {};
    int buf = 0;
    for (int k0 = 0; k0 < F_SZ; k0 += 32) {
        const int nxt = buf ^ 1;
        const bool has_next = (k0 + 32 < F_SZ);
        float4 br[2];
        if (has_next) {
            sel_stage_A(x, m0, k0 + 32, &Asf[nxt][0][0], tid);  // async DMA overlaps WMMA
            sel_load_B(Ws, n0, k0 + 32, br, tid);
        }

        // A fragment: fp32 LDS pairs -> bf16 (lane 0-15: K 0-7,16-23; 16-31: K 8-15,24-31)
        AFrag a;
        {
            int rl = wv * 16 + (lane & 15);
            int kh = (lane >> 4) * 8;
            const float* arow = &Asf[buf][rl][0];
            #pragma unroll
            for (int p = 0; p < 8; ++p) {
                int kk = ((p < 4) ? 2 * p : 16 + 2 * (p - 4)) + kh;
                float2 f2 = *(const float2*)&arow[kk];
                a.h[2 * p]     = to_bf16(f2.x);
                a.h[2 * p + 1] = to_bf16(f2.y);
            }
        }
        #pragma unroll
        for (int nt = 0; nt < 4; ++nt) {
            AFrag b;
            const unsigned* brow = (const unsigned*)&Bs[buf][nt * 16 + (lane & 15)][0];
            int kd = (lane >> 4) * 4;
            #pragma unroll
            for (int p = 0; p < 8; ++p) b.u[p] = brow[kd + ((p < 4) ? p : p + 4)];
            acc[nt] = __builtin_amdgcn_wmma_f32_16x16x32_bf16(
                false, a.v, false, b.v, (short)0, acc[nt], false, false);
        }

        if (has_next) sel_store_B(&Bs[nxt][0][0], br, tid);
#if HAS_ASYNC
        __builtin_amdgcn_s_wait_asynccnt(0);
#endif
        __syncthreads();
        buf = nxt;
    }

    // epilogue: + b_s, store S (row-major B x 8192)
    int mrow = wv * 16 + ((lane >> 4) << 3);
    int ncol = lane & 15;
    #pragma unroll
    for (int nt = 0; nt < 4; ++nt) {
        int n = n0 + nt * 16 + ncol;
        float bias = bs[n];
        #pragma unroll
        for (int v = 0; v < 8; ++v) {
            int m = m0 + mrow + v;
            S[m * NSEL + n] = acc[nt][v] + bias;
        }
    }
}

// ---------------------------------------------------------------------------
// Kernel 2: per (b,head) row of S: bitonic sort (desc) -> 4 thresholds ->
//           M = sigmoid((S - kth)/tau). One block (256 thr) per row.
// ---------------------------------------------------------------------------
__global__ __launch_bounds__(256)
void topk_masks(const float* __restrict__ S, const float* __restrict__ tau,
                float* __restrict__ M)
{
    __shared__ float s[1024];
    const int tid = threadIdx.x;
    const int row = blockIdx.x;                 // b*8 + head
    const float inv_tau = 1.0f / tau[0];

    *(float4*)&s[tid * 4] = *(const float4*)&S[row * F_SZ + tid * 4];
    __syncthreads();

    for (int ksz = 2; ksz <= 1024; ksz <<= 1) {
        for (int j = ksz >> 1; j > 0; j >>= 1) {
            #pragma unroll 2
            for (int t = tid; t < 512; t += 256) {
                int i   = 2 * t - (t & (j - 1));
                int ixj = i + j;
                float a = s[i], b = s[ixj];
                bool desc = ((i & ksz) == 0);
                bool sw   = desc ? (a < b) : (a > b);
                if (sw) { s[i] = b; s[ixj] = a; }
            }
            __syncthreads();
        }
    }

    const float th[4] = { s[31], s[63], s[127], s[255] };   // k = 32,64,128,256
    #pragma unroll
    for (int ki = 0; ki < 4; ++ki) {
        float t = th[ki];
        float4 sv = *(const float4*)&S[row * F_SZ + tid * 4];
        float4 mv;
        mv.x = 1.0f / (1.0f + __expf(-(sv.x - t) * inv_tau));
        mv.y = 1.0f / (1.0f + __expf(-(sv.y - t) * inv_tau));
        mv.z = 1.0f / (1.0f + __expf(-(sv.z - t) * inv_tau));
        mv.w = 1.0f / (1.0f + __expf(-(sv.w - t) * inv_tau));
        *(float4*)&M[(row * KSB + ki) * F_SZ + tid * 4] = mv;
    }
}

// ---------------------------------------------------------------------------
// Kernel 3: backbone. Rows r = b*32 + head*4 + ki (32768 total).
//   A[r][k] = x[b][k]*M[r][k] (bf16).  h = relu(A@W1 + b1) in LDS (bf16).
//   Y[r] = h@W2 + b2 (C padded to 112 = 7 WMMA n-tiles).
// block = 256 threads, 128 rows per block.
// ---------------------------------------------------------------------------
__global__ __launch_bounds__(256)
void backbone(const float* __restrict__ x,  const float* __restrict__ M,
              const float* __restrict__ W1, const float* __restrict__ b1,
              const float* __restrict__ W2, const float* __restrict__ b2,
              float* __restrict__ Y)
{
    __shared__ alignas(16) __bf16 As[128][32];     // (row, k)
    __shared__ alignas(16) __bf16 W1s[128][32];    // (n, k) col-major
    __shared__ alignas(16) __bf16 Hs[128][128];    // hidden, row-major

    const int tid  = threadIdx.x;
    const int lane = tid & 31;
    const int wv   = tid >> 5;
    const int r0   = blockIdx.x * 128;

    f32x8 acc[8] = {};
    for (int k0 = 0; k0 < F_SZ; k0 += 32) {
        // A = x*M, float4 loads, bf16x4 packed LDS stores
        #pragma unroll
        for (int i = 0; i < 4; ++i) {
            int f4 = tid + i * 256;                 // 0..1023
            int r = f4 >> 3, kq = f4 & 7;
            int gr = r0 + r, b = gr >> 5;           // 32 rows per batch elem
            float4 xv = *(const float4*)&x[b * F_SZ + k0 + kq * 4];
            float4 mv = *(const float4*)&M[gr * F_SZ + k0 + kq * 4];
            Pack4 pk;
            pk.h[0] = to_bf16(xv.x * mv.x); pk.h[1] = to_bf16(xv.y * mv.y);
            pk.h[2] = to_bf16(xv.z * mv.z); pk.h[3] = to_bf16(xv.w * mv.w);
            *(uint2*)&As[r][kq * 4] = pk.u2;
        }
        // W1 tile, float4 loads, transposed (n,k) stores
        #pragma unroll
        for (int i = 0; i < 4; ++i) {
            int f4 = tid + i * 256;                 // 0..1023
            int k = f4 >> 5, nq = f4 & 31;
            float4 wv4 = *(const float4*)&W1[(k0 + k) * H_SZ + nq * 4];
            W1s[nq * 4 + 0][k] = to_bf16(wv4.x);
            W1s[nq * 4 + 1][k] = to_bf16(wv4.y);
            W1s[nq * 4 + 2][k] = to_bf16(wv4.z);
            W1s[nq * 4 + 3][k] = to_bf16(wv4.w);
        }
        if (k0 + 32 < F_SZ)
            __builtin_prefetch(&M[(r0 + (tid >> 1)) * F_SZ + k0 + 32], 0, 0);
        __syncthreads();

        AFrag a;
        {
            int rl = wv * 16 + (lane & 15);
            int kh = (lane >> 4) * 8;
            const unsigned* arow = (const unsigned*)&As[rl][0];
            #pragma unroll
            for (int p = 0; p < 8; ++p) {
                int kk = ((p < 4) ? 2 * p : 16 + 2 * (p - 4)) + kh;
                a.u[p] = arow[kk >> 1];
            }
        }
        #pragma unroll
        for (int nt = 0; nt < 8; ++nt) {
            AFrag bf;
            const unsigned* brow = (const unsigned*)&W1s[nt * 16 + (lane & 15)][0];
            int kd = (lane >> 4) * 4;
            #pragma unroll
            for (int p = 0; p < 8; ++p) bf.u[p] = brow[kd + ((p < 4) ? p : p + 4)];
            acc[nt] = __builtin_amdgcn_wmma_f32_16x16x32_bf16(
                false, a.v, false, bf.v, (short)0, acc[nt], false, false);
        }
        __syncthreads();
    }

    // epilogue 1: bias + relu -> Hs (bf16)
    {
        int mrow = wv * 16 + ((lane >> 4) << 3);
        int ncol = lane & 15;
        #pragma unroll
        for (int nt = 0; nt < 8; ++nt) {
            int n = nt * 16 + ncol;
            float bias = b1[n];
            #pragma unroll
            for (int v = 0; v < 8; ++v) {
                float h = acc[nt][v] + bias;
                Hs[mrow + v][n] = to_bf16(h > 0.0f ? h : 0.0f);
            }
        }
    }
    __syncthreads();

    // GEMM2: (16x128) @ (128x112pad) per wave, K-steps of 32
    f32x8 acc2[7] = {};
    #pragma unroll
    for (int ks = 0; ks < 4; ++ks) {
        AFrag a;
        {
            int rl = wv * 16 + (lane & 15);
            int kh = (lane >> 4) * 8;
            const unsigned* arow = (const unsigned*)&Hs[rl][0];
            #pragma unroll
            for (int p = 0; p < 8; ++p) {
                int kk = ks * 32 + ((p < 4) ? 2 * p : 16 + 2 * (p - 4)) + kh;
                a.u[p] = arow[kk >> 1];
            }
        }
        #pragma unroll
        for (int nt = 0; nt < 7; ++nt) {
            AFrag bf;
            int col  = nt * 16 + (lane & 15);
            int colc = col < C_SZ ? col : (C_SZ - 1);
            float sc = col < C_SZ ? 1.0f : 0.0f;    // branchless pad: EXEC stays full
            #pragma unroll
            for (int p = 0; p < 8; ++p) {
                int k = ks * 32 + (lane >> 4) * 8 + ((p < 4) ? 2 * p : 16 + 2 * (p - 4));
                bf.h[2 * p]     = to_bf16(W2[k       * C_SZ + colc] * sc);
                bf.h[2 * p + 1] = to_bf16(W2[(k + 1) * C_SZ + colc] * sc);
            }
            acc2[nt] = __builtin_amdgcn_wmma_f32_16x16x32_bf16(
                false, a.v, false, bf.v, (short)0, acc2[nt], false, false);
        }
    }

    // epilogue 2: + b2, store Y (NROWS x 100)
    {
        int mrow = wv * 16 + ((lane >> 4) << 3);
        int ncol = lane & 15;
        #pragma unroll
        for (int nt = 0; nt < 7; ++nt) {
            int c = nt * 16 + ncol;
            if (c < C_SZ) {
                float bias = b2[c];
                #pragma unroll
                for (int v = 0; v < 8; ++v) {
                    int r = r0 + mrow + v;
                    Y[r * C_SZ + c] = acc2[nt][v] + bias;
                }
            }
        }
    }
}

// ---------------------------------------------------------------------------
// Kernel 4: final_probs[b] = mean_head softmax(Y[b, head, 3, :])
// one block (128 thr) per batch element
// ---------------------------------------------------------------------------
__global__ __launch_bounds__(128)
void softmax_mean(const float* __restrict__ Y, float* __restrict__ out)
{
    __shared__ float red[128];
    const int t = threadIdx.x;
    const int b = blockIdx.x;
    float pacc = 0.0f;

    for (int head = 0; head < KFC; ++head) {
        const float* yrow = Y + (size_t)((b * KFC + head) * KSB + (KSB - 1)) * C_SZ;
        float v = (t < C_SZ) ? yrow[t] : -3.0e38f;
        red[t] = v; __syncthreads();
        for (int s = 64; s > 0; s >>= 1) {
            if (t < s) red[t] = fmaxf(red[t], red[t + s]);
            __syncthreads();
        }
        float mx = red[0]; __syncthreads();
        float e = (t < C_SZ) ? __expf(v - mx) : 0.0f;
        red[t] = e; __syncthreads();
        for (int s = 64; s > 0; s >>= 1) {
            if (t < s) red[t] += red[t + s];
            __syncthreads();
        }
        float sum = red[0]; __syncthreads();
        pacc += e / sum;
    }
    if (t < C_SZ) out[b * C_SZ + t] = pacc * (1.0f / KFC);
}

// ---------------------------------------------------------------------------
extern "C" void kernel_launch(void* const* d_in, const int* in_sizes, int n_in,
                              void* d_out, int out_size, void* d_ws, size_t ws_size,
                              hipStream_t stream)
{
    const float* x   = (const float*)d_in[0];
    const float* tau = (const float*)d_in[1];
    const float* Ws  = (const float*)d_in[2];
    const float* bs  = (const float*)d_in[3];
    const float* W1  = (const float*)d_in[4];
    const float* b1  = (const float*)d_in[5];
    const float* W2  = (const float*)d_in[6];
    const float* b2  = (const float*)d_in[7];

    // d_out = concat(final_probs, Y, M, S) in return order (all fp32)
    float* out_final = (float*)d_out;                       // 1024*100
    float* out_Y = out_final + (size_t)B_SZ * C_SZ;         // 32768*100
    float* out_M = out_Y + (size_t)NROWS * C_SZ;            // 32768*1024
    float* out_S = out_M + (size_t)NROWS * F_SZ;            // 1024*8192

    selector_gemm<<<dim3(NSEL / 64, B_SZ / 128), 256, 0, stream>>>(x, Ws, bs, out_S);
    topk_masks<<<dim3(B_SZ * KFC), 256, 0, stream>>>(out_S, tau, out_M);
    backbone<<<dim3(NROWS / 128), 256, 0, stream>>>(x, out_M, W1, b1, W2, b2, out_Y);
    softmax_mean<<<dim3(B_SZ), 128, 0, stream>>>(out_Y, out_final);
}
